// Model_3745211482439
// MI455X (gfx1250) — matrix operand order repaired and checked by reading, not compiled, not run on hardware
//
#include <hip/hip_runtime.h>
#include <hip/hip_bf16.h>

#define Nn 10000
#define Ff 512
#define Hh 128

typedef __attribute__((ext_vector_type(2)))  float  v2f;
typedef __attribute__((ext_vector_type(8)))  float  v8f;
typedef __attribute__((ext_vector_type(8)))  __bf16 v8bf;
typedef __attribute__((ext_vector_type(16))) __bf16 v16bf;

static inline int cdiv(long long a, long long b) { return (int)((a + b - 1) / b); }

// ---------------- degree / normalization ----------------
__global__ void deg_kernel(const int* __restrict__ ei, float* __restrict__ deg, int E_) {
    int e = blockIdx.x * blockDim.x + threadIdx.x;
    if (e >= E_) return;
    atomicAdd(&deg[ei[E_ + e]], 1.0f);
}

__global__ void rsqrt_kernel(float* __restrict__ dis, int n) {
    int i = blockIdx.x * blockDim.x + threadIdx.x;
    if (i >= n) return;
    dis[i] = rsqrtf(dis[i] + 1.0f);   // +1 self loop
}

// ---------------- f32 WMMA GEMM: C = A[M,K] @ B[K,Nc] (+bias, relu) ----------------
// one 16x16 C tile per wave, V_WMMA_F32_16X16X4_F32.
// K/Nc compile-time so all addressing strength-reduces to 32-bit adds + imm offsets.
template <int K, int Nc, bool RELU, bool BIAS>
__global__ void gemm_f32_wmma(const float* __restrict__ A, const float* __restrict__ B,
                              const float* __restrict__ bias, float* __restrict__ C,
                              int M)
{
    int wave = blockIdx.x * (blockDim.x >> 5) + (threadIdx.x >> 5);
    int lane = threadIdx.x & 31;
    constexpr int ntiles = Nc >> 4;
    int mt = wave / ntiles;
    int nt = wave - mt * ntiles;
    if (mt >= (M >> 4)) return;           // uniform per wave
    int m0 = mt << 4, n0 = nt << 4;
    int g  = lane >> 4;                   // half-wave group
    int ml = lane & 15;

    const float* arow = A + (m0 + ml) * K + 2 * g;       // 32-bit index, < 2^31
    const float* bcol = B + n0 + ml + 2 * g * Nc;
    v8f c = {};
    #pragma unroll 8
    for (int k = 0; k < K; k += 4) {
        // A 16x4: lanes 0-15 -> K = k,k+1 ; lanes 16-31 -> K = k+2,k+3
        v2f a = *(const v2f*)(arow + k);
        // B 4x16: VGPR0 rows k (g=0) / k+2 (g=1); VGPR1 rows k+1 / k+3
        v2f b;
        b[0] = bcol[k * Nc];
        b[1] = bcol[k * Nc + Nc];
        c = __builtin_amdgcn_wmma_f32_16x16x4_f32(false, a, false, b, (short)0, c, false, false);
    }
    int col = n0 + ml;
    float bb = BIAS ? bias[col] : 0.0f;
    float* crow = C + (m0 + 8 * g) * Nc + col;
    #pragma unroll
    for (int r = 0; r < 8; ++r) {
        float v = c[r] + bb;
        if (RELU) v = fmaxf(v, 0.0f);
        crow[r * Nc] = v;
    }
}

// ---------------- edge scatter: acc[dst] += h[src] * dis[src]*dis[dst] ----------------
template <int W>
__global__ void scatter_edges(const float* __restrict__ h, const int* __restrict__ ei,
                              const float* __restrict__ dis, float* __restrict__ acc,
                              int E_)
{
    int idx = blockIdx.x * blockDim.x + threadIdx.x;
    constexpr int vpe = W >> 2;            // float4 chunks per edge
    if (idx >= E_ * vpe) return;
    int e = idx / vpe;
    int c = (idx - e * vpe) << 2;
    int s = ei[e], d = ei[E_ + e];
    float coef = dis[s] * dis[d];
    const float4 hv = *(const float4*)(h + s * W + c);
    float* ap = acc + d * W + c;
    atomicAdd(ap + 0, hv.x * coef);
    atomicAdd(ap + 1, hv.y * coef);
    atomicAdd(ap + 2, hv.z * coef);
    atomicAdd(ap + 3, hv.w * coef);
}

// ---------------- finalize: out = relu(acc + h*dis^2 + bias) ----------------
template <int W>
__global__ void finalize_gcn(const float* __restrict__ acc, const float* __restrict__ h,
                             const float* __restrict__ dis, const float* __restrict__ bias,
                             float* __restrict__ out, int n)
{
    int idx = blockIdx.x * blockDim.x + threadIdx.x;
    if (idx >= n * W) return;
    int i = idx / W, c = idx - i * W;
    float ds = dis[i];
    float v = acc[idx] + h[idx] * ds * ds + bias[c];
    out[idx] = fmaxf(v, 0.0f);
}

// ---------------- p = x_hat * x_hat ----------------
__global__ void square_kernel(const float* __restrict__ x, float* __restrict__ p, int n) {
    int i = blockIdx.x * blockDim.x + threadIdx.x;
    if (i >= n) return;
    float v = x[i];
    p[i] = v * v;
}

// ---------------- f32 -> bf16 ----------------
__global__ void cvt_bf16_kernel(const float* __restrict__ in, __bf16* __restrict__ out, int n) {
    int i = blockIdx.x * blockDim.x + threadIdx.x;
    if (i >= n) return;
    out[i] = (__bf16)in[i];
}

// ---------------- pack Wf[K=128, N] -> WfT_bf16[N, 128] (K contiguous) ----------------
__global__ void pack_wft_kernel(const float* __restrict__ Wf, __bf16* __restrict__ WfT) {
    int idx = blockIdx.x * blockDim.x + threadIdx.x;  // idx = k*Nn + n (coalesced read)
    if (idx >= Hh * Nn) return;
    int k = idx / Nn, n = idx - k * Nn;
    WfT[n * Hh + k] = (__bf16)Wf[idx];
}

// ---------------- struct = sigmoid(hmid_bf16 @ WfT_bf16^T + bf) ----------------
// V_WMMA_F32_16X16X32_BF16, K=128 (4 steps), 5 N-tiles per wave for A reuse.
// Sigmoid via v_exp_f32 + v_rcp_f32 (no IEEE-divide slow path) to stay under the
// 400MB-store roofline (~17us at 23.3 TB/s).
__global__ void linkpred_gemm_bf16(const __bf16* __restrict__ Abf, const __bf16* __restrict__ BT,
                                   const float* __restrict__ bias, float* __restrict__ out)
{
    const int NT = 5;
    int wave = blockIdx.x * (blockDim.x >> 5) + (threadIdx.x >> 5);
    int lane = threadIdx.x & 31;
    const int nstrips = (Nn / 16) / NT;   // 125
    int mt = wave / nstrips;
    int st = wave - mt * nstrips;
    if (mt >= Nn / 16) return;            // uniform per wave
    int m0 = mt * 16;
    int g  = lane >> 4;
    int ml = lane & 15;

    // A 16x32 bf16 per k-step: lane holds K=[8g,8g+8) and K=[16+8g,16+8g+8) chunks
    v16bf a[4];
    const __bf16* arow = Abf + (m0 + ml) * Hh;
    #pragma unroll
    for (int s = 0; s < 4; ++s) {
        v8bf lo = *(const v8bf*)(arow + 32 * s + 8 * g);
        v8bf hi = *(const v8bf*)(arow + 32 * s + 16 + 8 * g);
        a[s] = __builtin_shufflevector(lo, hi, 0,1,2,3,4,5,6,7,8,9,10,11,12,13,14,15);
    }

    for (int t = 0; t < NT; ++t) {
        int n0 = (st * NT + t) * 16;
        // B 32x16 bf16: lane (n=ml) holds K=[8g,8g+8) and K=[16+8g,16+8g+8) chunks of WfT row
        const __bf16* bcol = BT + (n0 + ml) * Hh + 8 * g;
        v8f c = {};
        #pragma unroll
        for (int s = 0; s < 4; ++s) {
            v8bf blo = *(const v8bf*)(bcol + 32 * s);
            v8bf bhi = *(const v8bf*)(bcol + 32 * s + 16);
            v16bf b = __builtin_shufflevector(blo, bhi, 0,1,2,3,4,5,6,7,8,9,10,11,12,13,14,15);
            c = __builtin_amdgcn_wmma_f32_16x16x32_bf16(false, a[s], false, b, (short)0, c, false, false);
        }
        int col = n0 + ml;
        float bb = bias[col];
        float* orow = out + (size_t)(m0 + 8 * g) * Nn + col;   // only place needing 64-bit
        #pragma unroll
        for (int r = 0; r < 8; ++r) {
            float v = c[r] + bb;
            orow[(size_t)r * Nn] = __builtin_amdgcn_rcpf(1.0f + __expf(-v));
        }
    }
}

extern "C" void kernel_launch(void* const* d_in, const int* in_sizes, int n_in,
                              void* d_out, int out_size, void* d_ws, size_t ws_size,
                              hipStream_t stream)
{
    const float* x   = (const float*)d_in[0];
    const int*   ei  = (const int*)d_in[1];
    const int    E_  = in_sizes[1] / 2;
    const float* We1 = (const float*)d_in[2];  const float* be1 = (const float*)d_in[3];
    const float* We2 = (const float*)d_in[4];  const float* be2 = (const float*)d_in[5];
    const float* Wd1 = (const float*)d_in[6];  const float* bd1 = (const float*)d_in[7];
    const float* Wd2 = (const float*)d_in[8];  const float* bd2 = (const float*)d_in[9];
    const float* Wl  = (const float*)d_in[10]; const float* bl  = (const float*)d_in[11];
    const float* Wf  = (const float*)d_in[12]; const float* bf  = (const float*)d_in[13];

    float* out_struct = (float*)d_out;                           // [N,N]
    float* out_xhat   = out_struct + (size_t)Nn * Nn;            // [N,F]
    float* out_z      = out_xhat   + (size_t)Nn * Ff;            // [N,H]

    // workspace carve
    char* w8 = (char*)d_ws;
    float* dis  = (float*)w8; w8 += (size_t)Nn * 4;
    float* bufA = (float*)w8; w8 += (size_t)Nn * Ff * 4;         // GEMM out (pre-agg), later p
    float* bufB = (float*)w8; w8 += (size_t)Nn * Ff * 4;         // agg accumulator, later hmid f32
    float* h1   = (float*)w8; w8 += (size_t)Nn * Hh * 4;
    float* d1   = (float*)w8; w8 += (size_t)Nn * Hh * 4;
    __bf16* WfT_bf  = (__bf16*)h1;   // aliases: h1 retired after layer-2 GEMM
    __bf16* hmid_bf = (__bf16*)d1;   // d1 retired after layer-4 GEMM

    const int T = 256;
    const int wpb = T / 32;

    // degree -> dis = rsqrt(deg+1)
    hipMemsetAsync(dis, 0, (size_t)Nn * 4, stream);
    deg_kernel<<<cdiv(E_, T), T, 0, stream>>>(ei, dis, E_);
    rsqrt_kernel<<<cdiv(Nn, T), T, 0, stream>>>(dis, Nn);

    const int wavesH = (Nn / 16) * (Hh / 16);   // Nc = 128
    const int wavesF = (Nn / 16) * (Ff / 16);   // Nc = 512

    // ---- encoder layer 1: h1 = relu(gcn(x, We1)) ----
    gemm_f32_wmma<Ff, Hh, false, false><<<cdiv(wavesH, wpb), T, 0, stream>>>(x, We1, nullptr, bufA, Nn);
    hipMemsetAsync(bufB, 0, (size_t)Nn * Hh * 4, stream);
    scatter_edges<Hh><<<cdiv((long long)E_ * (Hh / 4), T), T, 0, stream>>>(bufA, ei, dis, bufB, E_);
    finalize_gcn<Hh><<<cdiv((long long)Nn * Hh, T), T, 0, stream>>>(bufB, bufA, dis, be1, h1, Nn);

    // ---- encoder layer 2: z = relu(gcn(h1, We2)) ----
    gemm_f32_wmma<Hh, Hh, false, false><<<cdiv(wavesH, wpb), T, 0, stream>>>(h1, We2, nullptr, bufA, Nn);
    hipMemsetAsync(bufB, 0, (size_t)Nn * Hh * 4, stream);
    scatter_edges<Hh><<<cdiv((long long)E_ * (Hh / 4), T), T, 0, stream>>>(bufA, ei, dis, bufB, E_);
    finalize_gcn<Hh><<<cdiv((long long)Nn * Hh, T), T, 0, stream>>>(bufB, bufA, dis, be2, out_z, Nn);

    // ---- decoder layer 1: d1 = relu(gcn(z, Wd1)) ----
    gemm_f32_wmma<Hh, Hh, false, false><<<cdiv(wavesH, wpb), T, 0, stream>>>(out_z, Wd1, nullptr, bufA, Nn);
    hipMemsetAsync(bufB, 0, (size_t)Nn * Hh * 4, stream);
    scatter_edges<Hh><<<cdiv((long long)E_ * (Hh / 4), T), T, 0, stream>>>(bufA, ei, dis, bufB, E_);
    finalize_gcn<Hh><<<cdiv((long long)Nn * Hh, T), T, 0, stream>>>(bufB, bufA, dis, bd1, d1, Nn);

    // ---- decoder layer 2: x_hat = relu(gcn(d1, Wd2)) ----
    gemm_f32_wmma<Hh, Ff, false, false><<<cdiv(wavesF, wpb), T, 0, stream>>>(d1, Wd2, nullptr, bufA, Nn);
    hipMemsetAsync(bufB, 0, (size_t)Nn * Ff * 4, stream);
    scatter_edges<Ff><<<cdiv((long long)E_ * (Ff / 4), T), T, 0, stream>>>(bufA, ei, dis, bufB, E_);
    finalize_gcn<Ff><<<cdiv((long long)Nn * Ff, T), T, 0, stream>>>(bufB, bufA, dis, bd2, out_xhat, Nn);

    // ---- link predictor ----
    square_kernel<<<cdiv((long long)Nn * Ff, T), T, 0, stream>>>(out_xhat, bufA, Nn * Ff);
    // hmid = relu(p @ Wl + bl) -> bufB (f32)
    gemm_f32_wmma<Ff, Hh, true, true><<<cdiv(wavesH, wpb), T, 0, stream>>>(bufA, Wl, bl, bufB, Nn);
    cvt_bf16_kernel<<<cdiv((long long)Nn * Hh, T), T, 0, stream>>>(bufB, hmid_bf, Nn * Hh);
    pack_wft_kernel<<<cdiv((long long)Hh * Nn, T), T, 0, stream>>>(Wf, WfT_bf);

    // struct = sigmoid(hmid @ Wf + bf), bf16 WMMA
    int waves = (Nn / 16) * ((Nn / 16) / 5);
    linkpred_gemm_bf16<<<cdiv(waves, wpb), T, 0, stream>>>(hmid_bf, WfT_bf, bf, out_struct);
}
